// TwoStageAttentionLayerCrossSegments_31207232373408
// MI455X (gfx1250) — matrix operand order repaired and checked by reading, not compiled, not run on hardware
//
#include <hip/hip_runtime.h>
#include <hip/hip_bf16.h>
#include <math.h>

// ---------------------------------------------------------------------------
// Types for CDNA5 WMMA (gfx1250, wave32)
// ---------------------------------------------------------------------------
typedef __bf16 bf16_t;
typedef __attribute__((ext_vector_type(16))) __bf16 v16bf;
typedef __attribute__((ext_vector_type(2)))  __bf16 v2bf;
typedef __attribute__((ext_vector_type(8)))  float  v8f;
typedef __attribute__((ext_vector_type(4)))  unsigned u32x4;
typedef __attribute__((ext_vector_type(4)))  int      v4i_t;

#define D_MODEL 256
#define SEG_N   1024
#define H_N     8
#define DK_N    32
#define B_N     16
#define DFF_N   1024
#define NROWS   (B_N * 2 * SEG_N)   // 32768 total token rows

// ---------------------------------------------------------------------------
// CDNA5 async global->LDS path (ASYNCcnt), guarded so the build never breaks.
// Builtin prototype (from hipcc diagnostic): (v4i* global, v4i* lds, Ii, Ii)
// ---------------------------------------------------------------------------
#if defined(__AMDGCN__) && __has_builtin(__builtin_amdgcn_global_load_async_to_lds_b128) && \
    __has_builtin(__builtin_amdgcn_s_wait_asynccnt)
#define USE_ASYNC_LDS 1
#else
#define USE_ASYNC_LDS 0
#endif

#if USE_ASYNC_LDS
typedef __attribute__((address_space(1))) v4i_t* gp128_t;
typedef __attribute__((address_space(3))) v4i_t* lp128_t;
__device__ __forceinline__ void async_b128(void* lds, const void* g) {
    __builtin_amdgcn_global_load_async_to_lds_b128((gp128_t)(g), (lp128_t)(lds), 0, 0);
}
__device__ __forceinline__ void async_wait0() { __builtin_amdgcn_s_wait_asynccnt(0); }
#else
__device__ __forceinline__ void async_b128(void* lds, const void* g) {
    *(u32x4*)lds = *(const u32x4*)g;
}
__device__ __forceinline__ void async_wait0() {}
#endif

__device__ __forceinline__ unsigned short f32_to_bf16(float f) {
    union { float f; unsigned u; } v; v.f = f;
    unsigned r = v.u + 0x7FFFu + ((v.u >> 16) & 1u);   // round-to-nearest-even
    return (unsigned short)(r >> 16);
}

__device__ __forceinline__ v2bf upk2(unsigned u) {
    union { unsigned u; v2bf v; } x; x.u = u; return x.v;
}

// A fragment: 16x32 (MxK) bf16, ISA layout:
//   lane: m = lane&15, half = lane>>4
//   vgpr j (pair 2j,2j+1): K = (j&3)*2 + (j>>2)*16 + half*8  (+0,+1)
__device__ __forceinline__ v16bf load_frag_a(const unsigned short* A, int lda) {
    const int lane = threadIdx.x & 31;
    const int m = lane & 15, half = lane >> 4;
    const unsigned short* row = A + (size_t)m * lda;
    v16bf f;
#pragma unroll
    for (int j = 0; j < 8; ++j) {
        int k = ((j & 3) << 1) + ((j >> 2) << 4) + (half << 3);
        v2bf p = upk2(*(const unsigned*)(row + k));
        f[2 * j] = p[0]; f[2 * j + 1] = p[1];
    }
    return f;
}

// B fragment: 32x16 (KxN) bf16, supplied as BT (N x K, row-major):
//   lane: n = lane&15, khalf = lane>>4
//   vgpr j (pair): K = (j&3)*2 + (j>>2)*16 + khalf*8 (+0,+1)
__device__ __forceinline__ v16bf load_frag_b(const unsigned short* BT, int ldb) {
    const int lane = threadIdx.x & 31;
    const int n = lane & 15, kh = lane >> 4;
    const unsigned short* row = BT + (size_t)n * ldb + (kh << 3);
    v16bf f;
#pragma unroll
    for (int j = 0; j < 8; ++j) {
        v2bf p = upk2(*(const unsigned*)(row + ((j & 3) << 1) + ((j >> 2) << 4)));
        f[2 * j] = p[0]; f[2 * j + 1] = p[1];
    }
    return f;
}

__device__ __forceinline__ v8f wmma_bf16(v16bf a, v16bf b, v8f c) {
    return __builtin_amdgcn_wmma_f32_16x16x32_bf16(false, a, false, b,
                                                   (short)0, c, false, false);
}

// ---------------------------------------------------------------------------
// f32 -> bf16 conversion / transpose-conversion
// ---------------------------------------------------------------------------
__global__ void cvt_bf16_kernel(const float* __restrict__ in,
                                unsigned short* __restrict__ out, int n) {
    int i = blockIdx.x * 256 + threadIdx.x;
    if (i < n) out[i] = f32_to_bf16(in[i]);
}

// W: K x N row-major  ->  WT: N x K row-major (bf16)
__global__ void transpose_cvt_kernel(const float* __restrict__ W,
                                     unsigned short* __restrict__ WT,
                                     int K, int N) {
    int i = blockIdx.x * 256 + threadIdx.x;
    if (i < K * N) {
        int k = i / N, n = i - k * N;
        WT[(size_t)n * K + k] = f32_to_bf16(W[i]);
    }
}

// ---------------------------------------------------------------------------
// Tiled bf16 GEMM:  C[M,N] = A[M,K] (bf16) * BT[N,K]^T (bf16) + bias
// Block tile 128x64, 8 waves, each wave: 16 rows x 64 cols (4 WMMA C tiles).
// Tiles staged to LDS with GLOBAL_LOAD_ASYNC_TO_LDS_B128 (ASYNCcnt).
// EPI: 0 = store bf16, 1 = store f32, 2 = exact-GELU then bf16
// ---------------------------------------------------------------------------
template <int EPI>
__global__ __launch_bounds__(256) void gemm_bf16_kernel(
    const unsigned short* __restrict__ A, long long strideA,
    const unsigned short* __restrict__ BT,
    const float* __restrict__ bias,
    void* __restrict__ C, long long strideC,
    int M, int N, int K) {
    __shared__ unsigned short sA[128 * 32];
    __shared__ unsigned short sB[64 * 32];

    const int tid  = threadIdx.x;
    const int wave = tid >> 5;
    const int lane = tid & 31;
    const int nIdx = lane & 15, half = lane >> 4;
    const int m0 = blockIdx.y * 128;
    const int n0 = blockIdx.x * 64;
    const unsigned short* Ab = A + (size_t)blockIdx.z * strideA;

    v8f acc[4];
#pragma unroll
    for (int t = 0; t < 4; ++t) acc[t] = (v8f){0,0,0,0,0,0,0,0};

    for (int kb = 0; kb < K; kb += 32) {
        __syncthreads();   // previous iteration's fragment reads all retired
        // A tile 128x32: 512 x b128 chunks, 2 per thread, LDS-direct
#pragma unroll
        for (int i = 0; i < 2; ++i) {
            int idx = tid + i * 256;
            int r = idx >> 2, c4 = idx & 3;
            async_b128(sA + r * 32 + c4 * 8,
                       Ab + (size_t)(m0 + r) * K + kb + c4 * 8);
        }
        // BT tile 64x32: 256 x b128 chunks, 1 per thread
        {
            int r = tid >> 2, c4 = tid & 3;
            async_b128(sB + r * 32 + c4 * 8,
                       BT + (size_t)(n0 + r) * K + kb + c4 * 8);
        }
        // prefetch next A tile lines while async copies are in flight
        if (kb + 32 < K) {
            int r = tid >> 1;
            __builtin_prefetch(Ab + (size_t)(m0 + r) * K + kb + 32, 0, 0);
        }
        async_wait0();
        __syncthreads();

        v16bf a = load_frag_a(sA + wave * 16 * 32, 32);
#pragma unroll
        for (int t = 0; t < 4; ++t) {
            v16bf b = load_frag_b(sB + t * 16 * 32, 32);
            acc[t] = wmma_bf16(a, b, acc[t]);
        }
    }

    // epilogue: C layout -> row m = j + 8*half, col n = lane&15 (+16*t)
    const int rbase = m0 + wave * 16;
#pragma unroll
    for (int t = 0; t < 4; ++t) {
        int col = n0 + t * 16 + nIdx;
        float bv = bias[col];
#pragma unroll
        for (int j = 0; j < 8; ++j) {
            int row = rbase + j + 8 * half;
            float v = acc[t][j] + bv;
            size_t off = (size_t)blockIdx.z * strideC + (size_t)row * N + col;
            if (EPI == 1) {
                ((float*)C)[off] = v;
            } else if (EPI == 0) {
                ((unsigned short*)C)[off] = f32_to_bf16(v);
            } else {
                float g = 0.5f * v * (1.0f + erff(v * 0.70710678118654752f));
                ((unsigned short*)C)[off] = f32_to_bf16(g);
            }
        }
    }
}

// ---------------------------------------------------------------------------
// Flash attention, one block per (qtile, b*H+h, stage).
// stage 0: softmax(q k^T / sqrt(DK)) v ; stage 1: softmax(k q^T / sqrt(DK)) q
// Block = 8 waves; wave owns 16 q-rows; K/V streamed in 64-wide chunks via LDS.
// K chunk uses async-to-LDS; V is transposed in flight (manual path).
// ---------------------------------------------------------------------------
__global__ __launch_bounds__(256) void attn_flash_kernel(
    const unsigned short* __restrict__ qb,
    const unsigned short* __restrict__ kb,
    const unsigned short* __restrict__ vb,
    unsigned short* __restrict__ Out) {
    const int SC = 64;                     // seg chunk
    __shared__ unsigned short sK[64 * 32];      // K chunk rows (s,dk)  == BT for S
    __shared__ unsigned short sVT[32 * 64];     // V chunk transposed (dk,s) == BT for PV
    __shared__ unsigned short sP[8 * 16 * 64];  // per-wave P staging (16 x SC)

    const int tid  = threadIdx.x;
    const int wave = tid >> 5;
    const int lane = tid & 31;
    const int nIdx = lane & 15, half = lane >> 4;
    const int b = blockIdx.y >> 3;
    const int h = blockIdx.y & 7;
    const int stage = blockIdx.z;
    const int q0 = blockIdx.x * 128 + wave * 16;   // q-row base for this wave

    const size_t base = (size_t)b * SEG_N * D_MODEL + h * DK_N;
    const unsigned short* Qp = (stage ? kb : qb) + base;  // "query" matrix
    const unsigned short* Kp = (stage ? qb : kb) + base;  // "key" matrix
    const unsigned short* Vp = (stage ? qb : vb) + base;  // "value" matrix

    // q fragment (16 x DK), DK == WMMA K exactly
    v16bf aq = load_frag_a(Qp + (size_t)q0 * D_MODEL, D_MODEL);

    v8f o[2]; o[0] = (v8f){0,0,0,0,0,0,0,0}; o[1] = (v8f){0,0,0,0,0,0,0,0};
    float mstat[8], lstat[8];
#pragma unroll
    for (int j = 0; j < 8; ++j) { mstat[j] = -1e30f; lstat[j] = 0.0f; }

    const float scale = 0.17677669529663687f;  // 1/sqrt(32)
    unsigned short* pw = sP + wave * 16 * SC;

    for (int s0 = 0; s0 < SEG_N; s0 += SC) {
        __syncthreads();
        // K chunk (64x32): 256 x b128 chunks, 1 per thread, LDS-direct async
        {
            int r = tid >> 2, c4 = tid & 3;
            async_b128(sK + r * 32 + c4 * 8,
                       Kp + (size_t)(s0 + r) * D_MODEL + c4 * 8);
        }
        // V chunk transposed into LDS (element shuffle -> manual path)
#pragma unroll
        for (int i = 0; i < 4; ++i) {
            int idx = tid + i * 256;            // 1024 uints
            int r = idx >> 4, c2 = idx & 15;
            unsigned uV = *(const unsigned*)(Vp + (size_t)(s0 + r) * D_MODEL + c2 * 2);
            sVT[(c2 * 2 + 0) * SC + r] = (unsigned short)(uV & 0xFFFFu);
            sVT[(c2 * 2 + 1) * SC + r] = (unsigned short)(uV >> 16);
        }
        // prefetch next chunk's K/V lines
        if (s0 + SC < SEG_N) {
            int r = tid >> 2;
            __builtin_prefetch(Kp + (size_t)(s0 + SC + r) * D_MODEL, 0, 0);
            __builtin_prefetch(Vp + (size_t)(s0 + SC + r) * D_MODEL, 0, 0);
        }
        async_wait0();
        __syncthreads();

        // S tiles: q (16x32) x K_chunk^T -> 4 x (16x16)
        v8f st[4];
#pragma unroll
        for (int t = 0; t < 4; ++t) {
            v16bf bk = load_frag_b(sK + t * 16 * 32, 32);
            st[t] = wmma_bf16(aq, bk, (v8f){0,0,0,0,0,0,0,0});
        }

        // online softmax (row stats per vgpr j, consistent across half-wave)
        float alpha[8], rs[8];
#pragma unroll
        for (int j = 0; j < 8; ++j) {
            float v = fmaxf(fmaxf(st[0][j], st[1][j]), fmaxf(st[2][j], st[3][j])) * scale;
            v = fmaxf(v, __shfl_xor(v, 1, 32));
            v = fmaxf(v, __shfl_xor(v, 2, 32));
            v = fmaxf(v, __shfl_xor(v, 4, 32));
            v = fmaxf(v, __shfl_xor(v, 8, 32));
            float mn = fmaxf(mstat[j], v);
            alpha[j] = __expf(mstat[j] - mn);
            mstat[j] = mn;
            rs[j] = 0.0f;
        }
        // P = exp(S*scale - m), staged to LDS in bf16 (same-wave, DS in-order)
#pragma unroll
        for (int t = 0; t < 4; ++t) {
#pragma unroll
            for (int j = 0; j < 8; ++j) {
                float p = __expf(st[t][j] * scale - mstat[j]);
                rs[j] += p;
                pw[(size_t)(j + 8 * half) * SC + t * 16 + nIdx] = f32_to_bf16(p);
            }
        }
#pragma unroll
        for (int j = 0; j < 8; ++j) {
            float v = rs[j];
            v += __shfl_xor(v, 1, 32);
            v += __shfl_xor(v, 2, 32);
            v += __shfl_xor(v, 4, 32);
            v += __shfl_xor(v, 8, 32);
            lstat[j] = alpha[j] * lstat[j] + v;
            o[0][j] *= alpha[j];
            o[1][j] *= alpha[j];
        }
        // o += P (16xSC) x V_chunk (SCx32): contraction in 32-steps
#pragma unroll
        for (int kk = 0; kk < SC; kk += 32) {
            v16bf ap = load_frag_a(pw + kk, SC);
#pragma unroll
            for (int nt = 0; nt < 2; ++nt) {
                v16bf bv = load_frag_b(sVT + (size_t)nt * 16 * SC + kk, SC);
                o[nt] = wmma_bf16(ap, bv, o[nt]);
            }
        }
    }

    // normalize and write: concat row = stage*SEG + q0 + m ; col = h*32 + dk
    const size_t orow0 = (size_t)b * 2 * SEG_N + (size_t)stage * SEG_N + q0;
#pragma unroll
    for (int nt = 0; nt < 2; ++nt) {
#pragma unroll
        for (int j = 0; j < 8; ++j) {
            int row = j + 8 * half;
            float v = o[nt][j] / lstat[j];
            Out[(orow0 + row) * D_MODEL + h * DK_N + nt * 16 + nIdx] = f32_to_bf16(v);
        }
    }
}

// ---------------------------------------------------------------------------
// Residual + LayerNorm over D=256; one wave per row, 8 rows per block.
// Writes f32 (residual chain / output) and optional bf16 (next GEMM input).
// ---------------------------------------------------------------------------
__global__ __launch_bounds__(256) void resid_ln_kernel(
    const float* __restrict__ A, const float* __restrict__ Bsrc,
    const float* __restrict__ g, const float* __restrict__ beta,
    float* __restrict__ outF, unsigned short* __restrict__ outB, int nrows) {
    const int wave = threadIdx.x >> 5, lane = threadIdx.x & 31;
    const int row = blockIdx.x * 8 + wave;
    if (row >= nrows) return;
    const size_t base = (size_t)row * D_MODEL;

    float v[8], s = 0.0f;
#pragma unroll
    for (int i = 0; i < 8; ++i) {
        int c = lane + i * 32;
        v[i] = A[base + c] + Bsrc[base + c];
        s += v[i];
    }
#pragma unroll
    for (int m = 16; m >= 1; m >>= 1) s += __shfl_xor(s, m, 32);
    float mu = s * (1.0f / 256.0f);
    float var = 0.0f;
#pragma unroll
    for (int i = 0; i < 8; ++i) { float d = v[i] - mu; var += d * d; }
#pragma unroll
    for (int m = 16; m >= 1; m >>= 1) var += __shfl_xor(var, m, 32);
    float r = rsqrtf(var * (1.0f / 256.0f) + 1e-5f);
#pragma unroll
    for (int i = 0; i < 8; ++i) {
        int c = lane + i * 32;
        float o = (v[i] - mu) * r * g[c] + beta[c];
        outF[base + c] = o;
        if (outB) outB[base + c] = f32_to_bf16(o);
    }
}

// ---------------------------------------------------------------------------
// Launch
// ---------------------------------------------------------------------------
extern "C" void kernel_launch(void* const* d_in, const int* in_sizes, int n_in,
                              void* d_out, int out_size, void* d_ws, size_t ws_size,
                              hipStream_t stream) {
    (void)in_sizes; (void)n_in; (void)out_size; (void)ws_size;
    const float* x   = (const float*)d_in[0];
    const float* Wq  = (const float*)d_in[1];  const float* bq  = (const float*)d_in[2];
    const float* Wk  = (const float*)d_in[3];  const float* bk  = (const float*)d_in[4];
    const float* Wv  = (const float*)d_in[5];  const float* bv  = (const float*)d_in[6];
    const float* Wo  = (const float*)d_in[7];  const float* bo  = (const float*)d_in[8];
    const float* g31 = (const float*)d_in[9];  const float* b31 = (const float*)d_in[10];
    const float* W1  = (const float*)d_in[11]; const float* b1  = (const float*)d_in[12];
    const float* W2  = (const float*)d_in[13]; const float* b2  = (const float*)d_in[14];
    const float* g41 = (const float*)d_in[15]; const float* b41 = (const float*)d_in[16];
    float* out = (float*)d_out;

    char* ws = (char*)d_ws;
    size_t off = 0;
    auto alloc = [&](size_t bytes) -> void* {
        void* p = ws + off;
        off += (bytes + 255) & ~(size_t)255;
        return p;
    };

    unsigned short* xb  = (unsigned short*)alloc((size_t)NROWS * D_MODEL * 2);
    unsigned short* Wqt = (unsigned short*)alloc((size_t)D_MODEL * D_MODEL * 2);
    unsigned short* Wkt = (unsigned short*)alloc((size_t)D_MODEL * D_MODEL * 2);
    unsigned short* Wvt = (unsigned short*)alloc((size_t)D_MODEL * D_MODEL * 2);
    unsigned short* Wot = (unsigned short*)alloc((size_t)D_MODEL * D_MODEL * 2);
    unsigned short* W1t = (unsigned short*)alloc((size_t)D_MODEL * DFF_N * 2);
    unsigned short* W2t = (unsigned short*)alloc((size_t)DFF_N * D_MODEL * 2);
    unsigned short* qbf = (unsigned short*)alloc((size_t)B_N * SEG_N * D_MODEL * 2);
    unsigned short* kbf = (unsigned short*)alloc((size_t)B_N * SEG_N * D_MODEL * 2);
    unsigned short* vbf = (unsigned short*)alloc((size_t)B_N * SEG_N * D_MODEL * 2);
    unsigned short* ao  = (unsigned short*)alloc((size_t)NROWS * D_MODEL * 2);
    float*          yf  = (float*)alloc((size_t)NROWS * D_MODEL * 4);
    float*          hf  = (float*)alloc((size_t)NROWS * D_MODEL * 4);
    unsigned short* tb  = (unsigned short*)alloc((size_t)NROWS * DFF_N * 2);
    // buffer reuse (dead after their last reader):
    unsigned short* hb = ao;   // attn output dead once yf is computed
    float*          mf = yf;   // y f32 dead once first LayerNorm is done

    // 1) conversions
    {
        int n = NROWS * D_MODEL;
        cvt_bf16_kernel<<<(n + 255) / 256, 256, 0, stream>>>(x, xb, n);
        int nw = D_MODEL * D_MODEL;
        transpose_cvt_kernel<<<(nw + 255) / 256, 256, 0, stream>>>(Wq, Wqt, D_MODEL, D_MODEL);
        transpose_cvt_kernel<<<(nw + 255) / 256, 256, 0, stream>>>(Wk, Wkt, D_MODEL, D_MODEL);
        transpose_cvt_kernel<<<(nw + 255) / 256, 256, 0, stream>>>(Wv, Wvt, D_MODEL, D_MODEL);
        transpose_cvt_kernel<<<(nw + 255) / 256, 256, 0, stream>>>(Wo, Wot, D_MODEL, D_MODEL);
        int n1 = D_MODEL * DFF_N;
        transpose_cvt_kernel<<<(n1 + 255) / 256, 256, 0, stream>>>(W1, W1t, D_MODEL, DFF_N);
        transpose_cvt_kernel<<<(n1 + 255) / 256, 256, 0, stream>>>(W2, W2t, DFF_N, D_MODEL);
    }

    // 2) Q/K/V projections — only the halves that are consumed.
    //    Q from segment rows [0,1024), K/V from rows [1024,2048) per batch.
    {
        dim3 grid(D_MODEL / 64, SEG_N / 128, B_N);
        long long sa = (long long)2 * SEG_N * D_MODEL;   // batch stride in x rows
        long long sc = (long long)SEG_N * D_MODEL;
        gemm_bf16_kernel<0><<<grid, 256, 0, stream>>>(xb, sa, Wqt, bq, qbf, sc,
                                                      SEG_N, D_MODEL, D_MODEL);
        gemm_bf16_kernel<0><<<grid, 256, 0, stream>>>(xb + (size_t)SEG_N * D_MODEL, sa,
                                                      Wkt, bk, kbf, sc,
                                                      SEG_N, D_MODEL, D_MODEL);
        gemm_bf16_kernel<0><<<grid, 256, 0, stream>>>(xb + (size_t)SEG_N * D_MODEL, sa,
                                                      Wvt, bv, vbf, sc,
                                                      SEG_N, D_MODEL, D_MODEL);
    }

    // 3) two-stage flash attention -> ao (bf16, concat on seq dim)
    {
        dim3 grid(SEG_N / 128, B_N * H_N, 2);
        attn_flash_kernel<<<grid, 256, 0, stream>>>(qbf, kbf, vbf, ao);
    }

    // 4) y = ao @ Wo + bo (f32)
    {
        dim3 grid(D_MODEL / 64, NROWS / 128, 1);
        gemm_bf16_kernel<1><<<grid, 256, 0, stream>>>(ao, 0, Wot, bo, yf, 0,
                                                      NROWS, D_MODEL, D_MODEL);
    }

    // 5) h = LN(x + y) -> hf (f32) + hb (bf16)
    resid_ln_kernel<<<NROWS / 8, 256, 0, stream>>>(x, yf, g31, b31, hf, hb, NROWS);

    // 6) t = gelu(h @ W1 + b1) (bf16)
    {
        dim3 grid(DFF_N / 64, NROWS / 128, 1);
        gemm_bf16_kernel<2><<<grid, 256, 0, stream>>>(hb, 0, W1t, b1, tb, 0,
                                                      NROWS, DFF_N, D_MODEL);
    }

    // 7) mlp = t @ W2 + b2 (f32)
    {
        dim3 grid(D_MODEL / 64, NROWS / 128, 1);
        gemm_bf16_kernel<1><<<grid, 256, 0, stream>>>(tb, 0, W2t, b2, mf, 0,
                                                      NROWS, D_MODEL, DFF_N);
    }

    // 8) out = LN(h + mlp)
    resid_ln_kernel<<<NROWS / 8, 256, 0, stream>>>(hf, mf, g41, b41, out, nullptr, NROWS);
}
